// NormalizedFourierKernelAttention_15161234555379
// MI455X (gfx1250) — matrix operand order repaired and checked by reading, not compiled, not run on hardware
//
#include <hip/hip_runtime.h>

typedef _Float16 v16h __attribute__((ext_vector_type(16)));
typedef _Float16 v8h  __attribute__((ext_vector_type(8)));
typedef float    v8f  __attribute__((ext_vector_type(8)));
typedef unsigned int u32x4 __attribute__((ext_vector_type(4)));
typedef int          i32x8 __attribute__((ext_vector_type(8)));
typedef int          i32x4 __attribute__((ext_vector_type(4)));

#define H_   16
#define B_   8
#define N_   32
#define D_   64
#define S_   4096
#define NH_  1024
#define SPLIT 8
#define SCH  64                       // s-rows per TDM chunk
#define FREQ0 ((float)(3.14159265358979323846 * 32.0 / 31.0))

// ---------------- zero scratch kv ----------------
__global__ void k_zero(float* p, int n) {
    int i = blockIdx.x * blockDim.x + threadIdx.x;
    if (i < n) p[i] = 0.0f;
}

// ---------------- per-(h,b) max |q|,|k| ----------------
__global__ void k_maxq(const float* __restrict__ key, const float* __restrict__ query,
                       float* __restrict__ maxq) {
    int bh = blockIdx.x;            // bh = h*B + b
    int h = bh >> 3, b = bh & 7;
    int t = threadIdx.x;            // 256 threads
    float m = 0.0f;
    for (int s = t; s < S_; s += 256) {
        size_t base = ((size_t)(b * S_ + s)) * H_ + h;
        m = fmaxf(m, fmaxf(fabsf(key[base]), fabsf(query[base])));
    }
    __shared__ float red[256];
    red[t] = m;
    __syncthreads();
    for (int w = 128; w > 0; w >>= 1) {
        if (t < w) red[t] = fmaxf(red[t], red[t + w]);
        __syncthreads();
    }
    if (t == 0) maxq[bh] = red[0];
}

// ---------------- spline evaluation: ra/ia per (h,b,n) ----------------
__global__ void k_spline(const float* __restrict__ ra_in, const float* __restrict__ ia_in,
                         const float* __restrict__ sc_in, const float* __restrict__ maxq,
                         float* __restrict__ ra_out, float* __restrict__ ia_out) {
    int bh = blockIdx.x;
    int h = bh >> 3;
    int n = threadIdx.x;            // 32 threads, one per knot
    float x = sc_in[h];
    float sa = ((x > 20.0f) ? x : log1pf(__expf(x))) + 0.5f;   // softplus + 0.5
    float mq = maxq[bh];
    const float step = 1.0f / 31.0f;
    float xx = n * step * mq;
    float accr = 0.0f, acci = 0.0f;
    for (int j = 0; j < 31; ++j) {
        float a0 = j * step * sa;
        float a1 = (j + 1) * step * sa;
        bool seg = (xx >= a0) && (xx < a1);
        float r0 = ra_in[h * 31 + j];
        float r1 = (j == 30) ? 0.0f : ra_in[h * 31 + j + 1];
        float i0 = ia_in[h * 31 + j];
        float i1 = (j == 30) ? 0.0f : ia_in[h * 31 + j + 1];
        float tt = (xx - a0) / (a1 - a0);
        if (seg) { accr += r0 + tt * (r1 - r0); acci += i0 + tt * (i1 - i0); }
    }
    ra_out[bh * 32 + n] = accr;
    ia_out[bh * 32 + n] = acci;
}

// ---------------- TDM descriptors: 64x64 f32 tile, global row stride NH_ ----------------
// D# per CDNA5 ISA ch.8: group0 = {flags, lds_addr, gaddr_lo, gaddr_hi|type},
// group1 = {mask/data_size, dims/tiles/strides}. 2-D tensor -> groups 2/3 zero.
__device__ __forceinline__ void tdm_build(const float* gptr, unsigned lds_off,
                                          u32x4& g0, i32x8& g1) {
    unsigned long long ga = (unsigned long long)(uintptr_t)gptr;
    g0[0] = 1u;                                             // count=1, user mode
    g0[1] = lds_off;                                        // lds_addr (bytes)
    g0[2] = (unsigned)(ga & 0xFFFFFFFFu);                   // global_addr[31:0]
    g0[3] = (unsigned)((ga >> 32) & 0x01FFFFFFu) | (2u << 30);  // global_addr[56:32] | type=2
    g1[0] = 0x20000;      // workgroup_mask=0, data_size=2 (4 bytes)
    g1[1] = 0x00400000;   // tensor_dim0 = 64  (bits 79:48, low 16 here)
    g1[2] = 0x00400000;   // tensor_dim0 hi = 0 | tensor_dim1 = 64 (low 16 in bits 95:80)
    g1[3] = 0x00400000;   // tensor_dim1 hi = 0 | tile_dim0 = 64 (bits 127:112)
    g1[4] = 64;           // tile_dim1 = 64, tile_dim2 = 0
    g1[5] = 1024;         // tensor_dim0_stride = NH_ (bits 207:160, low 32)
    g1[6] = 0;
    g1[7] = 0;
}

__device__ __forceinline__ void tdm_load_tile(const float* gptr, unsigned lds_off) {
    u32x4 g0; i32x8 g1;
    tdm_build(gptr, lds_off, g0, g1);
    i32x4 gz4 = {0, 0, 0, 0};
    i32x8 gz8 = {0, 0, 0, 0, 0, 0, 0, 0};
    __builtin_amdgcn_tensor_load_to_lds(g0, g1, gz4, gz4, gz8, 0);
}

__device__ __forceinline__ void tdm_store_tile(const float* gptr, unsigned lds_off) {
    u32x4 g0; i32x8 g1;
    tdm_build(gptr, lds_off, g0, g1);
    i32x4 gz4 = {0, 0, 0, 0};
    i32x8 gz8 = {0, 0, 0, 0, 0, 0, 0, 0};
    __builtin_amdgcn_tensor_store_from_lds(g0, g1, gz4, gz4, gz8, 0);
}

// ---------------- GEMM1: kv[f][d] = sum_s trig(freq[n]*k[s]) * v[s][d] ----------------
// grid (SPLIT, 128), 128 threads = 4 waves. wave w owns f-rows [16w,16w+16).
// V tiles land in LDS via the Tensor Data Mover, double buffered.
__global__ void __launch_bounds__(128) k_kv(const float* __restrict__ key,
                                            const float* __restrict__ value,
                                            const float* __restrict__ maxq,
                                            float* __restrict__ kv) {
    int bh = blockIdx.y;
    int h = bh >> 3, b = bh & 7;
    int split = blockIdx.x;
    int tid = threadIdx.x, lane = tid & 31, wave = tid >> 5;

    __shared__ __align__(16) float raw[2][SCH][64];     // TDM landing buffers (16KB x2)
    __shared__ __align__(16) _Float16 Vt[64][80];       // [d][s_local] f16, padded rows
    __shared__ float kn[SCH];                           // raw key chunk (un-normalized)

    float inv = 1.0f / maxq[bh];
    v8f acc[4] = {};

    int f = wave * 16 + (lane & 15);      // A-matrix row M (ISA: M = lane&15)
    int n = f & 31;
    bool is_sin = (f >= 32);
    float freq = n * FREQ0 * inv;         // normalization folded into frequency
    int kbase = (lane < 16) ? 0 : 8;      // ISA 16-bit A layout K interleave
    int khalf = lane >> 4;

    const int sbeg = split * (S_ / SPLIT);
    const int NCH = (S_ / SPLIT) / SCH;   // 8 chunks of 64 rows
    const size_t gstride = (size_t)NH_;

    unsigned lds0 = (unsigned)(uintptr_t)(void*)&raw[0][0][0];
    unsigned lds1 = (unsigned)(uintptr_t)(void*)&raw[1][0][0];

    if (wave == 0)
        tdm_load_tile(value + ((size_t)(b * S_ + sbeg)) * gstride + h * D_, lds0);

    for (int c = 0; c < NCH; ++c) {
        int s0 = sbeg + c * SCH;
        int cur = c & 1;
        if (wave == 0) {
            if (c + 1 < NCH) {
                tdm_load_tile(value + ((size_t)(b * S_ + s0 + SCH)) * gstride + h * D_,
                              cur ? lds0 : lds1);
                __builtin_amdgcn_s_wait_tensorcnt(1);   // current tile done
            } else {
                __builtin_amdgcn_s_wait_tensorcnt(0);
            }
        }
        if (tid < SCH)
            kn[tid] = key[((size_t)(b * S_ + s0 + tid)) * H_ + h];
        __syncthreads();            // raw[cur] + kn visible

        // --- transpose + cvt pass: raw f32 [s][d] -> Vt f16 [d][s] ---
        {
            int sl = tid >> 1, dh = tid & 1;
            const float* src = &raw[cur][sl][dh * 32];
#pragma unroll
            for (int j = 0; j < 32; j += 4) {
                float4 v4 = *(const float4*)(src + j);
                Vt[dh * 32 + j + 0][sl] = (_Float16)v4.x;
                Vt[dh * 32 + j + 1][sl] = (_Float16)v4.y;
                Vt[dh * 32 + j + 2][sl] = (_Float16)v4.z;
                Vt[dh * 32 + j + 3][sl] = (_Float16)v4.w;
            }
        }
        __syncthreads();            // Vt ready

        // --- 2 K-chunks of 32: A from trig in regs, B from LDS, 8 WMMAs/wave ---
#pragma unroll
        for (int ck = 0; ck < 2; ++ck) {
            v16h a;
#pragma unroll
            for (int e = 0; e < 16; ++e) {
                int K = 32 * ck + kbase + e + ((e >= 8) ? 8 : 0);
                float arg = freq * kn[K];
                a[e] = (_Float16)(is_sin ? __sinf(arg) : __cosf(arg));
            }
#pragma unroll
            for (int nt = 0; nt < 4; ++nt) {
                int d = nt * 16 + (lane & 15);
                const v8h* p = (const v8h*)&Vt[d][ck * 32 + khalf * 8];
                v8h b0 = p[0], b1 = p[2];
                v16h bf;
#pragma unroll
                for (int e = 0; e < 8; ++e) { bf[e] = b0[e]; bf[e + 8] = b1[e]; }
                acc[nt] = __builtin_amdgcn_wmma_f32_16x16x32_f16(
                    false, a, false, bf, (short)0, acc[nt], false, false);
            }
        }
        __syncthreads();            // before Vt is overwritten
    }

    // --- accumulate partial kv into global scratch (C/D layout per ISA) ---
#pragma unroll
    for (int nt = 0; nt < 4; ++nt) {
#pragma unroll
        for (int r = 0; r < 8; ++r) {
            int fr = wave * 16 + r + 8 * (lane >> 4);
            int d = nt * 16 + (lane & 15);
            atomicAdd(&kv[(size_t)bh * 4096 + fr * 64 + d], acc[nt][r]);
        }
    }
}

// ---------------- GEMM2: out[q][d] = sum_f A(q,f) * kv[f][d] ----------------
// grid (64, 128), 128 threads = 4 waves; WG covers 64 q-rows of batch bh.
// Output tile staged in LDS and written by the TDM (linear full-line HBM writes).
__global__ void __launch_bounds__(128) k_out(const float* __restrict__ query,
                                             const float* __restrict__ kv,
                                             const float* __restrict__ maxq,
                                             const float* __restrict__ ra,
                                             const float* __restrict__ ia,
                                             float* __restrict__ out) {
    int bh = blockIdx.y;
    int h = bh >> 3, b = bh & 7;
    int qt = blockIdx.x;
    int tid = threadIdx.x, lane = tid & 31, wave = tid >> 5;

    __shared__ __align__(16) _Float16 kvh[64][80];  // [d][f], padded rows
    __shared__ __align__(16) float oTile[64][64];   // [q_local][d] staging for TDM store
    __shared__ float ras[32], ias[32], qn[64];

    float inv = 1.0f / maxq[bh];

    // stage kv (64x64 f32) transposed to f16 LDS
    {
        int fr = tid >> 1, dh = tid & 1;
        const float* src = kv + (size_t)bh * 4096 + fr * 64 + dh * 32;
#pragma unroll
        for (int j = 0; j < 32; j += 4) {
            float4 v4 = *(const float4*)(src + j);
            kvh[dh * 32 + j + 0][fr] = (_Float16)v4.x;
            kvh[dh * 32 + j + 1][fr] = (_Float16)v4.y;
            kvh[dh * 32 + j + 2][fr] = (_Float16)v4.z;
            kvh[dh * 32 + j + 3][fr] = (_Float16)v4.w;
        }
        if (tid < 32) { ras[tid] = ra[bh * 32 + tid]; ias[tid] = ia[bh * 32 + tid]; }
        if (tid < 64)
            qn[tid] = query[((size_t)(b * S_ + qt * 64 + tid)) * H_ + h] * inv;
    }
    __syncthreads();

    int m = lane & 15;
    float qv = qn[wave * 16 + m];
    int kbase = (lane < 16) ? 0 : 8;
    int khalf = lane >> 4;

    // A fragments for K-chunks 0 (vs kv_c) and 1 (vs kv_s); trig shared per knot n
    v16h a0, a1;
#pragma unroll
    for (int e = 0; e < 16; ++e) {
        int n = kbase + e + ((e >= 8) ? 8 : 0);
        float arg = n * FREQ0 * qv;
        float sn = __sinf(arg), cs = __cosf(arg);
        float r_ = ras[n], i_ = ias[n];
        a0[e] = (_Float16)(r_ * cs + i_ * sn);   // coefficient of kv_c
        a1[e] = (_Float16)(r_ * sn - i_ * cs);   // coefficient of kv_s
    }

#pragma unroll
    for (int nt = 0; nt < 4; ++nt) {
        int d = nt * 16 + (lane & 15);
        const v8h* p0 = (const v8h*)&kvh[d][khalf * 8];
        const v8h* p1 = (const v8h*)&kvh[d][32 + khalf * 8];
        v8h c0 = p0[0], c1 = p0[2], s0 = p1[0], s1 = p1[2];
        v16h bf0, bf1;
#pragma unroll
        for (int e = 0; e < 8; ++e) {
            bf0[e] = c0[e]; bf0[e + 8] = c1[e];
            bf1[e] = s0[e]; bf1[e + 8] = s1[e];
        }
        v8f acc = {};
        acc = __builtin_amdgcn_wmma_f32_16x16x32_f16(false, a0, false, bf0, (short)0, acc, false, false);
        acc = __builtin_amdgcn_wmma_f32_16x16x32_f16(false, a1, false, bf1, (short)0, acc, false, false);
#pragma unroll
        for (int r = 0; r < 8; ++r) {
            int ql = wave * 16 + r + 8 * (lane >> 4);
            oTile[ql][d] = acc[r];
        }
    }
    __syncthreads();    // full 64x64 output tile staged

    if (wave == 0) {
        tdm_store_tile(out + ((size_t)(b * S_ + qt * 64)) * NH_ + h * D_,
                       (unsigned)(uintptr_t)(void*)&oTile[0][0]);
        __builtin_amdgcn_s_wait_tensorcnt(0);   // LDS stays alive until DMA done
    }
}

// ---------------- launch ----------------
extern "C" void kernel_launch(void* const* d_in, const int* in_sizes, int n_in,
                              void* d_out, int out_size, void* d_ws, size_t ws_size,
                              hipStream_t stream) {
    const float* key   = (const float*)d_in[0];
    const float* value = (const float*)d_in[1];
    const float* query = (const float*)d_in[2];
    const float* ra_in = (const float*)d_in[3];
    const float* ia_in = (const float*)d_in[4];
    const float* sc_in = (const float*)d_in[5];
    float* out = (float*)d_out;

    float* maxq = (float*)d_ws;        // 128
    float* ra   = maxq + 128;          // 128*32
    float* ia   = ra + 128 * 32;       // 128*32
    float* kvws = ia + 128 * 32;       // 128*64*64 = 524288 floats

    const int kvN = 128 * 64 * 64;
    k_zero<<<(kvN + 255) / 256, 256, 0, stream>>>(kvws, kvN);
    k_maxq<<<128, 256, 0, stream>>>(key, query, maxq);
    k_spline<<<128, 32, 0, stream>>>(ra_in, ia_in, sc_in, maxq, ra, ia);
    k_kv<<<dim3(SPLIT, 128), 128, 0, stream>>>(key, value, maxq, kvws);
    k_out<<<dim3(64, 128), 128, 0, stream>>>(query, kvws, maxq, ra, ia, out);
}
